// CPAlayer_14499809591710
// MI455X (gfx1250) — matrix operand order repaired and checked by reading, not compiled, not run on hardware
//
#include <hip/hip_runtime.h>
#include <hip/hip_bf16.h>

typedef __attribute__((ext_vector_type(16))) _Float16 v16h;
typedef __attribute__((ext_vector_type(8)))  float    v8f;

#define BATCH 4
#define NROW  4096
#define DIM   512

// Branch-free tanh: native v_tanh_f32 if the toolchain has it, else
// tanh(u) = 1 - 2/(exp(2u)+1) via native v_exp_f32 + v_rcp_f32.
__device__ __forceinline__ float fast_tanh(float u) {
#if __has_builtin(__builtin_amdgcn_tanhf)
  return __builtin_amdgcn_tanhf(u);
#else
  float e = __expf(2.0f * u);
  return 1.0f - 2.0f * __builtin_amdgcn_rcpf(e + 1.0f);
#endif
}

// g(z) = exp(leakyrelu_0.1(8*tanh(z/8)))
__device__ __forceinline__ float g_act(float z) {
  float t = fast_tanh(z * 0.125f) * 8.0f;
  float l = (t >= 0.0f) ? t : 0.1f * t;   // v_cndmask, branch-free
  return __expf(l);
}

__device__ __forceinline__ v8f wmma16(v16h a, v16h b, v8f c) {
  return __builtin_amdgcn_wmma_f32_16x16x32_f16(false, a, false, b, (short)0, c,
                                                false, false);
}

// ---------------- stage 1: v = w @ a  (512 floats) ----------------
__global__ __launch_bounds__(256) void k_compute_v(const float* __restrict__ w,
                                                   const float* __restrict__ a,
                                                   float* __restrict__ v) {
  int di = blockIdx.x * 256 + threadIdx.x;
  if (di < DIM) {
    float sum = 0.0f;
    const float* wr = w + (size_t)di * DIM;
    #pragma unroll 8
    for (int k = 0; k < DIM; ++k) sum += wr[k] * a[k];
    v[di] = sum;
  }
}

// ---------------- stage 2: s[b,n] = x[b,n,:] . v  (one wave per row) --------
__global__ __launch_bounds__(256) void k_compute_s(const float* __restrict__ x,
                                                   const float* __restrict__ v,
                                                   float* __restrict__ s) {
  int row  = blockIdx.x * 8 + (threadIdx.x >> 5);   // 8 waves per block
  int lane = threadIdx.x & 31;
  const float* xr = x + (size_t)row * DIM;
  float sum = 0.0f;
  #pragma unroll
  for (int k = lane; k < DIM; k += 32) sum += xr[k] * v[k];
  #pragma unroll
  for (int o = 16; o > 0; o >>= 1) sum += __shfl_xor(sum, o, 32);
  if (lane == 0) s[row] = sum;
}

// ---------------- stage 3a: pack w -> f16 B-fragment layout ----------------
// wh[kk2(16)][ct(32)][lane(32)][h(16)] ; K = kk2*32 + h + 16*(lane>>4),
// col = ct*16 + (lane&15); B-frag = one contiguous v16h per lane.
__global__ __launch_bounds__(256) void k_pack_w(const float* __restrict__ w,
                                                _Float16* __restrict__ wh) {
  int e    = blockIdx.x * 256 + threadIdx.x;        // 262144 total
  int h    = e & 15;
  int lane = (e >> 4) & 31;
  int ct   = (e >> 9) & 31;
  int kk   = e >> 14;
  int K    = kk * 32 + h + 16 * (lane >> 4);
  int col  = ct * 16 + (lane & 15);
  wh[e] = (_Float16)w[(size_t)K * DIM + col];
}

// ---------------- stage 3b: pack x -> f16 B-fragment layout ----------------
// xh[b][kk(128)][ct(32)][lane(32)][h(16)]; j = kk*32 + h + 16*(lane>>4),
// d = ct*16 + (lane&15)
__global__ __launch_bounds__(256) void k_pack_x(const float* __restrict__ x,
                                                _Float16* __restrict__ xh) {
  size_t e = (size_t)blockIdx.x * 256 + threadIdx.x; // 8388608 total
  int h    = (int)(e & 15);
  int lane = (int)(e >> 4) & 31;
  int ct   = (int)(e >> 9) & 31;
  int kk   = (int)(e >> 14) & 127;
  int b    = (int)(e >> 21);
  int j    = kk * 32 + h + 16 * (lane >> 4);
  int d    = ct * 16 + (lane & 15);
  xh[e] = (_Float16)x[((size_t)b * NROW + j) * DIM + d];
}

// ---------------- fused main kernel ----------------
// One workgroup (256 thr = 8 waves) owns 32 rows of one batch, loops K=4096.
// Computes U = (g o A_shape) @ x  (WMMA f16, f32 acc) while accumulating the
// row sums r = sum g*As and d = sum As; then out_tile = (coef*U) @ w via a
// second WMMA GEMM through LDS.
__global__ __launch_bounds__(256) void k_fused(const float* __restrict__ As,
                                               const float* __restrict__ s,
                                               const _Float16* __restrict__ xh,
                                               const _Float16* __restrict__ wh,
                                               float* __restrict__ out) {
  __shared__ __align__(32) _Float16 lds_afrag[2][2 * 32 * 16]; // parity dbl-buf
  __shared__ __align__(32) _Float16 lds_uh[2 * 16 * 32 * 16];  // 32x512 A-frags
  __shared__ float lds_srow[32];
  __shared__ float lds_rs[32];
  __shared__ float lds_ds[32];
  __shared__ float lds_coef[32];

  const int b    = blockIdx.x >> 7;            // 128 row-blocks per batch
  const int i0   = (blockIdx.x & 127) * 32;
  const int t    = threadIdx.x;
  const int lane = t & 31;
  const int wave = t >> 5;                     // 0..7
  const int tr   = wave & 1;                   // 16-row tile within the 32 rows
  const int cg   = wave >> 1;                  // 128-col group (0..3)

  if (t < 32) {
    lds_srow[t] = s[b * NROW + i0 + t];
    lds_rs[t] = 0.0f;
    lds_ds[t] = 0.0f;
  }
  __syncthreads();

  // transform-thread assignment: row ti (0..31), cols tj..tj+3 of each K tile
  const int   ti = t >> 3;
  const int   tj = (t & 7) * 4;
  const float si = lds_srow[ti];
  const int   trI   = ti >> 4;
  const int   m16I  = ti & 15;
  float rsum = 0.0f, dsum = 0.0f;

  const float*    As_row = As + ((size_t)b * NROW + (i0 + ti)) * NROW;
  const float*    s_b    = s + b * NROW;
  const _Float16* xp     = xh + (size_t)b * (128 * 32 * 32 * 16);

  v8f acc[8] = {};

  for (int kk = 0; kk < 128; ++kk) {
    const int k0 = kk * 32;
    const int p  = kk & 1;

    // ---- transform: val = g(s_i + s_j) * A_shape, swizzled into A-frag LDS
    float4 a4 = *(const float4*)(As_row + k0 + tj);
    float4 s4 = *(const float4*)(s_b + k0 + tj);
    __builtin_prefetch(As_row + k0 + 32 + tj, 0, 1); // next tile -> global_prefetch
    float av[4] = {a4.x, a4.y, a4.z, a4.w};
    float sv[4] = {s4.x, s4.y, s4.z, s4.w};
    #pragma unroll
    for (int c = 0; c < 4; ++c) {
      float val = g_act(si + sv[c]) * av[c];
      rsum += val;
      dsum += av[c];
      int K  = tj + c;                       // 0..31 within tile
      int g  = K >> 4;
      int h  = K & 15;
      int lI = m16I + 16 * g;
      lds_afrag[p][(trI * 32 + lI) * 16 + h] = (_Float16)val;
    }
    __syncthreads();

    // ---- compute: 8 WMMAs per wave (16 rows x 128 cols, K=32)
    v16h afr = *(const v16h*)&lds_afrag[p][(tr * 32 + lane) * 16];
    #pragma unroll
    for (int u = 0; u < 8; ++u) {
      int  ct  = cg * 8 + u;
      v16h bfr = *(const v16h*)(xp + (((size_t)kk * 32 + ct) * 32 + lane) * 16);
      acc[u] = wmma16(afr, bfr, acc[u]);
    }
    // no second barrier needed: next iteration writes the other parity buffer
  }
  __syncthreads();

  // ---- row-sum reduction -> coef = degsum / rowsum
  atomicAdd(&lds_rs[ti], rsum);
  atomicAdd(&lds_ds[ti], dsum);
  __syncthreads();
  if (t < 32) lds_coef[t] = lds_ds[t] / lds_rs[t];
  __syncthreads();

  // ---- scale U by coef, store as f16 A-fragments (32 rows x 512 K2) in LDS
  const int lg = lane >> 4;
  const int ln = lane & 15;
  float cf[8];
  #pragma unroll
  for (int r = 0; r < 8; ++r) cf[r] = lds_coef[tr * 16 + r + 8 * lg];
  #pragma unroll
  for (int u = 0; u < 8; ++u) {
    int colb = cg * 128 + u * 16;
    #pragma unroll
    for (int r = 0; r < 8; ++r) {
      float val = acc[u][r] * cf[r];
      int col   = colb + ln;
      int kk2   = col >> 5;
      int Kin   = col & 31;
      int g2    = Kin >> 4;
      int h2    = Kin & 15;
      int lane2 = (r + 8 * lg) + 16 * g2;
      lds_uh[((tr * 16 + kk2) * 32 + lane2) * 16 + h2] = (_Float16)val;
    }
  }
  __syncthreads();

  // ---- second GEMM: out_tile = Uh(32x512) @ w(512x512), K2 = 512
  v8f acc2[8] = {};
  for (int k2 = 0; k2 < 16; ++k2) {
    v16h afr = *(const v16h*)&lds_uh[((tr * 16 + k2) * 32 + lane) * 16];
    #pragma unroll
    for (int u = 0; u < 8; ++u) {
      int  ct  = cg * 8 + u;
      v16h bfr = *(const v16h*)(wh + (((size_t)k2 * 32 + ct) * 32 + lane) * 16);
      acc2[u] = wmma16(afr, bfr, acc2[u]);
    }
  }

  // ---- write out: C layout VGPR r -> M = r + 8*(lane>>4), N = lane&15
  float* outp = out + ((size_t)b * NROW + i0) * DIM;
  #pragma unroll
  for (int u = 0; u < 8; ++u) {
    int colb = cg * 128 + u * 16;
    #pragma unroll
    for (int r = 0; r < 8; ++r) {
      int m = tr * 16 + r + 8 * lg;
      outp[(size_t)m * DIM + colb + ln] = acc2[u][r];
    }
  }
}

extern "C" void kernel_launch(void* const* d_in, const int* in_sizes, int n_in,
                              void* d_out, int out_size, void* d_ws, size_t ws_size,
                              hipStream_t stream) {
  const float* d_x  = (const float*)d_in[0];   // [4,4096,512]
  const float* d_As = (const float*)d_in[1];   // [4,4096,4096]
  const float* d_w  = (const float*)d_in[2];   // [512,512]
  const float* d_a  = (const float*)d_in[3];   // [512]
  float*       d_o  = (float*)d_out;

  char* ws = (char*)d_ws;
  float*    d_v  = (float*)(ws + 0);                       // 2 KB
  float*    d_s  = (float*)(ws + 2048);                    // 64 KB
  _Float16* d_wh = (_Float16*)(ws + 2048 + 65536);         // 512 KB
  _Float16* d_xh = (_Float16*)(ws + 2048 + 65536 + 524288); // 16 MB

  k_compute_v<<<2, 256, 0, stream>>>(d_w, d_a, d_v);
  k_compute_s<<<(BATCH * NROW) / 8, 256, 0, stream>>>(d_x, d_v, d_s);
  k_pack_w<<<(16 * 32 * 32 * 16) / 256, 256, 0, stream>>>(d_w, d_wh);
  k_pack_x<<<(BATCH * 128 * 32 * 32 * 16) / 256, 256, 0, stream>>>(d_x, d_xh);
  k_fused<<<BATCH * (NROW / 32), 256, 0, stream>>>(d_As, d_s, d_xh, d_wh, d_o);
}
